// RoIGAtR_24910810316996
// MI455X (gfx1250) — matrix operand order repaired and checked by reading, not compiled, not run on hardware
//
#include <hip/hip_runtime.h>
#include <cstdint>
#include <cstddef>

// ---------------- problem constants (from reference) ----------------
#define NN_ 4096      // nodes
#define IN_ 1024      // input features
#define RR_ 256       // out channels per head
#define HH_ 8         // heads (layer 1)
#define KK_ 6         // kNN
#define ES_ (NN_*KK_ + NN_)   // edge slots incl. self loops
#define NEG_SLOPE 0.2f

typedef __attribute__((ext_vector_type(16))) __bf16 v16bf;
typedef __attribute__((ext_vector_type(8)))  float  v8f;
typedef int v4i_ __attribute__((vector_size(16)));   // matches builtin param type

// -------- CDNA5 async global->LDS copy (ASYNCcnt path), guarded --------
#if __has_builtin(__builtin_amdgcn_global_load_async_to_lds_b128)
#define HAVE_ASYNC_LDS 1
#define COPY16_ASYNC(g, l)                                            \
  __builtin_amdgcn_global_load_async_to_lds_b128(                     \
      (__attribute__((address_space(1))) v4i_*)(g),                   \
      (__attribute__((address_space(3))) v4i_*)(l), 0, 0)
#else
#define COPY16_ASYNC(g, l) (*(uint4*)(l) = *(const uint4*)(g))
#endif

#if __has_builtin(__builtin_amdgcn_s_wait_asynccnt)
#define WAIT_ASYNC() __builtin_amdgcn_s_wait_asynccnt(0)
#elif defined(HAVE_ASYNC_LDS)
#define WAIT_ASYNC() asm volatile("s_wait_asynccnt 0x0" ::: "memory")
#else
#define WAIT_ASYNC() ((void)0)
#endif

// ---------------- fp32 -> bf16 (RNE) ----------------
__global__ void cvt_bf16_kernel(const float* __restrict__ in,
                                unsigned short* __restrict__ out, int n) {
  int idx = blockIdx.x * blockDim.x + threadIdx.x;
  if (idx < n) {
    unsigned int u = __float_as_uint(in[idx]);
    out[idx] = (unsigned short)((u + 0x7FFFu + ((u >> 16) & 1u)) >> 16);
  }
}

// ---------------- WMMA GEMM (double-buffered LDS, async staging) ----------------
// C[M,Nn] = A[M,Kd](bf16) * B(bf16) (+bias), f32 accum.
// BT=false: B is [Kd,Nn] row-major (x @ W).  BT=true: B is [Nn,Kd] row-major (x @ x^T).
// Requires M%64==0, Nn%128==0, Kd%32==0 (true for all call sites here).
template<bool BT>
__global__ __launch_bounds__(256)
void wmma_gemm_kernel(const __bf16* __restrict__ A, const __bf16* __restrict__ B,
                      const float* __restrict__ bias, float* __restrict__ C,
                      int M, int Nn, int Kd) {
  __shared__ __bf16 As[2][64][40];    // 64x32 tile, +8 pad, double buffered
  __shared__ __bf16 Bs[2][32][136];   // 32x128 tile (k-major), +8 pad, double buffered

  const int tid  = threadIdx.x;
  const int wave = tid >> 5, lane = tid & 31;
  const int hi   = lane >> 4, lrow = lane & 15;
  const int waveM = wave >> 2, waveN = wave & 3;   // 2x4 wave grid
  const int m0 = blockIdx.y * 64, n0 = blockIdx.x * 128;

  v8f acc[2][2] = {};

  auto stage = [&](int buf, int k0) {
    // ---- A tile: 64x32, one 16B chunk per thread (async DMA when available) ----
    {
      int idx = tid * 8, r = idx >> 5, kk = idx & 31;
      COPY16_ASYNC(&A[(size_t)(m0 + r) * Kd + k0 + kk], &As[buf][r][kk]);
    }
    // ---- B tile into Bs[k][n] ----
    if (!BT) {
      #pragma unroll
      for (int c2 = 0; c2 < 2; ++c2) {
        int idx = (tid + c2 * 256) * 8, k = idx >> 7, nn = idx & 127;
        COPY16_ASYNC(&B[(size_t)(k0 + k) * Nn + n0 + nn], &Bs[buf][k][nn]);
      }
    } else {
      // transpose staging must go through VGPRs (sync path)
      int n = tid & 127, kh = tid >> 7;            // 16 k-values per thread
      union { uint4 q[2]; __bf16 e[16]; } tb;
      const uint4* gp = (const uint4*)&B[(size_t)(n0 + n) * Kd + k0 + kh * 16];
      tb.q[0] = gp[0]; tb.q[1] = gp[1];
      #pragma unroll
      for (int e2 = 0; e2 < 16; ++e2) Bs[buf][kh * 16 + e2][n] = tb.e[e2];
      if (k0 + 32 < Kd)   // keep next K-slice near
        __builtin_prefetch(&B[(size_t)(n0 + n) * Kd + k0 + 32], 0, 3);
    }
  };

  const int nK = Kd >> 5;
  stage(0, 0);

  for (int kt = 0; kt < nK; ++kt) {
    const int buf = kt & 1;
    WAIT_ASYNC();          // this wave's DMA into `buf` complete
    __syncthreads();       // all waves' writes to `buf` visible; prev reads of buf^1 done
    if (kt + 1 < nK) stage(buf ^ 1, (kt + 1) * 32);   // overlap DMA with WMMA below

    // ---- build fragments per CDNA5 16-bit VGPR layouts ----
    v16bf afr[2], bfr[2];
    #pragma unroll
    for (int am = 0; am < 2; ++am) {
      int mr = waveM * 32 + am * 16 + lrow;
      #pragma unroll
      for (int v = 0; v < 8; ++v) {
        int kb = ((v < 4) ? (v * 2) : (16 + (v - 4) * 2)) + hi * 8;
        afr[am][2 * v]     = As[buf][mr][kb];
        afr[am][2 * v + 1] = As[buf][mr][kb + 1];
      }
    }
    #pragma unroll
    for (int bn = 0; bn < 2; ++bn) {
      int nc = waveN * 32 + bn * 16;
      #pragma unroll
      for (int e2 = 0; e2 < 16; ++e2) bfr[bn][e2] = Bs[buf][(e2 < 8 ? e2 : e2 + 8) + hi * 8][nc + lrow];
    }
    // ---- 4 WMMAs per wave per K-step ----
    #pragma unroll
    for (int am = 0; am < 2; ++am)
      #pragma unroll
      for (int bn = 0; bn < 2; ++bn)
        acc[am][bn] = __builtin_amdgcn_wmma_f32_16x16x32_bf16(
            false, afr[am], false, bfr[bn], (short)0, acc[am][bn], false, false);
  }

  // ---- epilogue: C/D layout -> global ----
  #pragma unroll
  for (int am = 0; am < 2; ++am)
    #pragma unroll
    for (int bn = 0; bn < 2; ++bn) {
      int col = n0 + waveN * 32 + bn * 16 + lrow;
      float bv = bias ? bias[col] : 0.0f;
      #pragma unroll
      for (int v = 0; v < 8; ++v) {
        int row = m0 + waveM * 32 + am * 16 + hi * 8 + v;
        C[(size_t)row * Nn + col] = acc[am][bn][v] + bv;
      }
    }
}

// ---------------- top-K per row of adj (block-per-row, coalesced) ----------------
__device__ __forceinline__ void topk_insert(float* bv, int* bi, float v, int c) {
  const int last = KK_ - 1;
  if (!((v > bv[last]) || (v == bv[last] && c < bi[last]))) return;
  int p = last;
  while (p > 0 && ((v > bv[p - 1]) || (v == bv[p - 1] && c < bi[p - 1]))) {
    bv[p] = bv[p - 1]; bi[p] = bi[p - 1]; --p;
  }
  bv[p] = v; bi[p] = c;
}

__global__ __launch_bounds__(256)
void topk_kernel(const float* __restrict__ adj,
                 float* __restrict__ topv, int* __restrict__ topi) {
  const int i = blockIdx.x;
  const int t = threadIdx.x;
  __shared__ float sv[256 * KK_];
  __shared__ int   si[256 * KK_];

  float bv[KK_]; int bi[KK_];
  #pragma unroll
  for (int j = 0; j < KK_; ++j) { bv[j] = -__builtin_inff(); bi[j] = 0x7fffffff; }

  const float* row = adj + (size_t)i * NN_;
  for (int c = t; c < NN_; c += 256)            // coalesced across lanes
    topk_insert(bv, bi, row[c], c);

  #pragma unroll
  for (int j = 0; j < KK_; ++j) { sv[t * KK_ + j] = bv[j]; si[t * KK_ + j] = bi[j]; }
  __syncthreads();

  if (t < 32) {                                  // merge 8 lists each
    float mv[KK_]; int mi[KK_];
    #pragma unroll
    for (int j = 0; j < KK_; ++j) { mv[j] = -__builtin_inff(); mi[j] = 0x7fffffff; }
    for (int s = t; s < 256; s += 32)
      #pragma unroll
      for (int j = 0; j < KK_; ++j)
        topk_insert(mv, mi, sv[s * KK_ + j], si[s * KK_ + j]);
    __syncthreads();
    #pragma unroll
    for (int j = 0; j < KK_; ++j) { sv[t * KK_ + j] = mv[j]; si[t * KK_ + j] = mi[j]; }
  } else {
    __syncthreads();
  }
  __syncthreads();

  if (t == 0) {                                  // final merge of 32 lists
    float fv[KK_]; int fi[KK_];
    #pragma unroll
    for (int j = 0; j < KK_; ++j) { fv[j] = -__builtin_inff(); fi[j] = 0x7fffffff; }
    for (int s = 0; s < 32; ++s)
      #pragma unroll
      for (int j = 0; j < KK_; ++j)
        topk_insert(fv, fi, sv[s * KK_ + j], si[s * KK_ + j]);
    #pragma unroll
    for (int j = 0; j < KK_; ++j) { topv[i * KK_ + j] = fv[j]; topi[i * KK_ + j] = fi[j]; }
  }
}

// ---------------- per-(edge,head) attention logit (wave32 dot) ----------------
// slot e < N*K: dst=e/K, src=topi[e], valid iff src>dst && topv!=0  (triu + remove_self_loops)
// slot e >= N*K: self loop (always valid)
__global__ void edge_alpha_kernel(const float* __restrict__ xl, const float* __restrict__ xr,
                                  const int* __restrict__ topi, const float* __restrict__ topv,
                                  const float* __restrict__ att, float* __restrict__ alpha,
                                  int Hh, int Cc) {
  int wid  = blockIdx.x * (blockDim.x >> 5) + (threadIdx.x >> 5);
  int lane = threadIdx.x & 31;
  int total = ES_ * Hh;
  if (wid >= total) return;
  int e = wid / Hh, h = wid - e * Hh;
  int src, dst; bool valid = true;
  if (e < NN_ * KK_) {
    dst = e / KK_; src = topi[e];
    valid = (src > dst) && (topv[e] != 0.0f);
  } else {
    dst = e - NN_ * KK_; src = dst;
  }
  int per = Cc >> 5;                       // 8 channels per lane (Cc=256)
  int c0 = lane * per;
  const float* pl = xl + ((size_t)src * Hh + h) * Cc + c0;
  const float* pr = xr + ((size_t)dst * Hh + h) * Cc + c0;
  const float* pa = att + (size_t)h * Cc + c0;
  float s = 0.0f;
  #pragma unroll 8
  for (int c = 0; c < per; ++c) {
    float v = pl[c] + pr[c];
    v = (v >= 0.0f) ? v : NEG_SLOPE * v;   // leaky_relu before attention dot (GATv2)
    s += v * pa[c];
  }
  #pragma unroll
  for (int off = 16; off; off >>= 1) s += __shfl_xor(s, off, 32);
  if (lane == 0) alpha[wid] = valid ? s : -__builtin_inff();
}

// ---------------- per-node segment softmax + aggregation + bias + PReLU ----------------
__global__ void aggregate_kernel(const float* __restrict__ xl, const float* __restrict__ alpha,
                                 const int* __restrict__ topi, const float* __restrict__ bias,
                                 const float* __restrict__ prelu_a, float* __restrict__ out,
                                 int Hh, int Cc) {
  int i = blockIdx.x;
  int t = threadIdx.x;
  __shared__ int   ssrc[KK_ + 1];
  __shared__ float swn[(KK_ + 1) * 8];
  if (t <= KK_) ssrc[t] = (t < KK_) ? topi[i * KK_ + t] : i;
  if (t < Hh) {
    float av[KK_ + 1];
    float m = -__builtin_inff();
    #pragma unroll
    for (int j = 0; j <= KK_; ++j) {
      int e = (j < KK_) ? (i * KK_ + j) : (NN_ * KK_ + i);
      av[j] = alpha[(size_t)e * Hh + t];
      m = fmaxf(m, av[j]);
    }
    float den = 0.0f;
    #pragma unroll
    for (int j = 0; j <= KK_; ++j) { float w = __expf(av[j] - m); av[j] = w; den += w; }
    float inv = 1.0f / den;
    #pragma unroll
    for (int j = 0; j <= KK_; ++j) swn[j * 8 + t] = av[j] * inv;
  }
  __syncthreads();
  float a = prelu_a[0];
  int total = Hh * Cc;
  for (int o = t; o < total; o += blockDim.x) {
    int h = o / Cc;
    float acc = 0.0f;
    #pragma unroll
    for (int j = 0; j <= KK_; ++j)
      acc += swn[j * 8 + h] * xl[(size_t)ssrc[j] * total + o];
    acc += bias[o];
    out[(size_t)i * total + o] = (acc >= 0.0f) ? acc : a * acc;
  }
}

// ---------------- launch ----------------
extern "C" void kernel_launch(void* const* d_in, const int* in_sizes, int n_in,
                              void* d_out, int out_size, void* d_ws, size_t ws_size,
                              hipStream_t stream) {
  const float* x     = (const float*)d_in[0];
  const float* w1_l  = (const float*)d_in[1];
  const float* b1_l  = (const float*)d_in[2];
  const float* w1_r  = (const float*)d_in[3];
  const float* b1_r  = (const float*)d_in[4];
  const float* att1  = (const float*)d_in[5];
  const float* bias1 = (const float*)d_in[6];
  const float* w2_l  = (const float*)d_in[7];
  const float* b2_l  = (const float*)d_in[8];
  const float* w2_r  = (const float*)d_in[9];
  const float* b2_r  = (const float*)d_in[10];
  const float* att2  = (const float*)d_in[11];
  const float* bias2 = (const float*)d_in[12];
  const float* a1    = (const float*)d_in[13];
  const float* a2    = (const float*)d_in[14];
  float* out = (float*)d_out;

  // workspace carve-out (256B aligned)
  size_t off = 0;
  auto carve = [&](size_t bytes) -> void* {
    void* p = (char*)d_ws + off;
    off += (bytes + 255) & ~(size_t)255;
    return p;
  };
  const int HR = HH_ * RR_;                       // 2048
  unsigned short* xbf   = (unsigned short*)carve((size_t)NN_ * IN_ * 2);
  unsigned short* w1lbf = (unsigned short*)carve((size_t)IN_ * HR * 2);
  unsigned short* w1rbf = (unsigned short*)carve((size_t)IN_ * HR * 2);
  unsigned short* w2lbf = (unsigned short*)carve((size_t)HR * RR_ * 2);
  unsigned short* w2rbf = (unsigned short*)carve((size_t)HR * RR_ * 2);
  unsigned short* h1bf  = (unsigned short*)carve((size_t)NN_ * HR * 2);
  float* adj    = (float*)carve((size_t)NN_ * NN_ * 4);
  float* topv   = (float*)carve((size_t)NN_ * KK_ * 4);
  int*   topi   = (int*)  carve((size_t)NN_ * KK_ * 4);
  float* xl1    = (float*)carve((size_t)NN_ * HR * 4);
  float* xr1    = (float*)carve((size_t)NN_ * HR * 4);
  float* alpha1 = (float*)carve((size_t)ES_ * HH_ * 4);
  float* h1     = (float*)carve((size_t)NN_ * HR * 4);
  float* xl2    = (float*)carve((size_t)NN_ * RR_ * 4);
  float* xr2    = (float*)carve((size_t)NN_ * RR_ * 4);
  float* alpha2 = (float*)carve((size_t)ES_ * 4);
  (void)ws_size; (void)in_sizes; (void)n_in; (void)out_size;

  auto cvt = [&](const float* src, unsigned short* dst, int n) {
    cvt_bf16_kernel<<<(n + 255) / 256, 256, 0, stream>>>(src, dst, n);
  };

  // 1) bf16 conversions for GEMM operands
  cvt(x,    xbf,   NN_ * IN_);
  cvt(w1_l, w1lbf, IN_ * HR);
  cvt(w1_r, w1rbf, IN_ * HR);
  cvt(w2_l, w2lbf, HR * RR_);
  cvt(w2_r, w2rbf, HR * RR_);

  // 2) adjacency: adj = x @ x^T   (WMMA, NT)
  wmma_gemm_kernel<true><<<dim3(NN_ / 128, NN_ / 64), 256, 0, stream>>>(
      (const __bf16*)xbf, (const __bf16*)xbf, nullptr, adj, NN_, NN_, IN_);

  // 3) top-K per row (diag is max -> self captured, masked as in reference)
  topk_kernel<<<NN_, 256, 0, stream>>>(adj, topv, topi);

  // 4) layer-1 linears: xl1 = x@w1_l + b1_l ; xr1 = x@w1_r + b1_r
  wmma_gemm_kernel<false><<<dim3(HR / 128, NN_ / 64), 256, 0, stream>>>(
      (const __bf16*)xbf, (const __bf16*)w1lbf, b1_l, xl1, NN_, HR, IN_);
  wmma_gemm_kernel<false><<<dim3(HR / 128, NN_ / 64), 256, 0, stream>>>(
      (const __bf16*)xbf, (const __bf16*)w1rbf, b1_r, xr1, NN_, HR, IN_);

  // 5) attention logits + softmax/aggregate (+bias1, PReLU a1) -> h1
  {
    int waves = ES_ * HH_;
    edge_alpha_kernel<<<(waves + 7) / 8, 256, 0, stream>>>(xl1, xr1, topi, topv,
                                                           att1, alpha1, HH_, RR_);
    aggregate_kernel<<<NN_, 256, 0, stream>>>(xl1, alpha1, topi, bias1, a1, h1, HH_, RR_);
  }

  // 6) layer-2 linears on h1
  cvt(h1, h1bf, NN_ * HR);
  wmma_gemm_kernel<false><<<dim3(RR_ / 128, NN_ / 64), 256, 0, stream>>>(
      (const __bf16*)h1bf, (const __bf16*)w2lbf, b2_l, xl2, NN_, RR_, HR);
  wmma_gemm_kernel<false><<<dim3(RR_ / 128, NN_ / 64), 256, 0, stream>>>(
      (const __bf16*)h1bf, (const __bf16*)w2rbf, b2_r, xr2, NN_, RR_, HR);

  // 7) layer-2 attention + aggregate (+bias2, PReLU a2) -> d_out
  {
    int waves = ES_ * 1;
    edge_alpha_kernel<<<(waves + 7) / 8, 256, 0, stream>>>(xl2, xr2, topi, topv,
                                                           att2, alpha2, 1, RR_);
    aggregate_kernel<<<NN_, 256, 0, stream>>>(xl2, alpha2, topi, bias2, a2, out, 1, RR_);
  }
}
